// GQA_31756988186703
// MI455X (gfx1250) — compile-verified, matrix operand order checked by reading, not run on hardware
//
#include <hip/hip_runtime.h>
#include <hip/hip_bf16.h>

// ---------------------------------------------------------------------------
// GQA forward for B=2, L=2048, DIM=1024, H=16, KV=4, HD=64 on gfx1250.
// bf16 WMMA for all matmuls; weights pre-transposed so every WMMA fragment is
// two contiguous b128 loads; flash attention with async global->LDS double
// buffering of K/V tiles (ASYNCcnt) and LDS relayout of P. wave32 throughout.
// ---------------------------------------------------------------------------

typedef __bf16 bf16_t;
typedef __attribute__((ext_vector_type(16))) __bf16 v16bf;
typedef __attribute__((ext_vector_type(8)))  __bf16 v8bf;
typedef __attribute__((ext_vector_type(8)))  float  v8f;

constexpr int Bc   = 2;
constexpr int Lc   = 2048;
constexpr int DIMc = 1024;
constexpr int Hc   = 16;
constexpr int KVc  = 4;
constexpr int HDc  = 64;
constexpr int Gc   = Hc / KVc;          // 4
constexpr int ROWS = Bc * Lc;           // 4096

// ---------------- fragment loader (wave32 WMMA VGPR layouts) ----------------
// 16-bit A-matrix 16x32 (MxK), per ISA 7.12.2:
//   lanes 0-15  : row M = lane,    K = {0..7} in v0..3, {16..23} in v4..7
//   lanes 16-31 : row M = lane-16, K = {8..15},        {24..31}
// A B-fragment (32x16 KxN) of a row-major matrix W equals the A-style
// fragment of W^T, so with transposed operands this is the only loader.
__device__ __forceinline__ v16bf load_a_frag(const bf16_t* __restrict__ p, int ld) {
  const int lane = threadIdx.x & 31;
  const int row  = lane & 15;
  const int kb   = (lane >> 4) << 3;     // 0 or 8
  const v8bf lo = *(const v8bf*)(p + row * ld + kb);
  const v8bf hi = *(const v8bf*)(p + row * ld + kb + 16);
  v16bf a;
#pragma unroll
  for (int i = 0; i < 8; ++i) { a[i] = lo[i]; a[8 + i] = hi[i]; }
  return a;
}

__device__ __forceinline__ v8f wmma_bf16(v16bf a, v16bf b, v8f c) {
  return __builtin_amdgcn_wmma_f32_16x16x32_bf16(
      /*neg_a=*/false, a, /*neg_b=*/false, b,
      /*c_mod=*/(short)0, c, /*reuse_a=*/false, /*reuse_b=*/false);
}

// --------------------------- fp32 -> bf16 convert ---------------------------
__global__ void cvt_bf16_kernel(const float* __restrict__ src,
                                bf16_t* __restrict__ dst, int n) {
  int i = blockIdx.x * blockDim.x + threadIdx.x;
  if (i < n) dst[i] = (bf16_t)src[i];
}

// fp32 [rows, cols] -> bf16 [cols, rows]  (one-time weight transpose)
__global__ void cvt_bf16_t_kernel(const float* __restrict__ src,
                                  bf16_t* __restrict__ dst, int rows, int cols) {
  int idx = blockIdx.x * blockDim.x + threadIdx.x;
  if (idx >= rows * cols) return;
  int r = idx / cols, c = idx - r * cols;
  dst[(size_t)c * rows + r] = (bf16_t)src[idx];
}

// ------------------------------ WMMA GEMM (TN) ------------------------------
// C[M,N] f32 = A[M,K] bf16 (row-major) x BT[N,K] bf16 (row-major, = B^T).
// 4 waves/block; wave w owns M-tile (blockIdx.y*4+w); each wave computes a
// 16x64 C strip: one A fragment feeds 4 WMMAs per K-step.
__global__ __launch_bounds__(128)
void gemm_bf16_tn_kernel(const bf16_t* __restrict__ A, const bf16_t* __restrict__ BT,
                         float* __restrict__ C, int M, int N, int K) {
  const int wave = threadIdx.x >> 5;
  const int lane = threadIdx.x & 31;
  const int mt = blockIdx.y * 4 + wave;
  const int n0 = blockIdx.x * 64;
  const bf16_t* a_tile = A + (size_t)mt * 16 * K;

  v8f acc[4] = {};
  for (int k0 = 0; k0 < K; k0 += 32) {
    if (k0 + 64 < K) __builtin_prefetch(a_tile + (lane & 15) * K + k0 + 64, 0, 0);
    const v16bf a = load_a_frag(a_tile + k0, K);
#pragma unroll
    for (int t = 0; t < 4; ++t) {
      const v16bf b = load_a_frag(BT + (size_t)(n0 + t * 16) * K + k0, K);
      acc[t] = wmma_bf16(a, b, acc[t]);
    }
  }
  // C/D layout: VGPR r holds row (r + 8*(lane>>4)), col = lane&15 within tile.
  const int col   = n0 + (lane & 15);
  const int rbase = mt * 16 + ((lane >> 4) << 3);
#pragma unroll
  for (int t = 0; t < 4; ++t)
#pragma unroll
    for (int r = 0; r < 8; ++r)
      C[(size_t)(rbase + r) * N + col + t * 16] = acc[t][r];
}

// ---------------------- RoPE + per-head bf16 repack -------------------------
// proj: f32 [B*L, nh*64] ; out: bf16 [B, nh, L, 64]
__global__ void rope_pack_kernel(const float* __restrict__ proj,
                                 const float* __restrict__ fcos,
                                 const float* __restrict__ fsin,
                                 bf16_t* __restrict__ out, int nh) {
  int idx = blockIdx.x * blockDim.x + threadIdx.x;
  int total = Bc * Lc * nh * (HDc / 2);
  if (idx >= total) return;
  int pr = idx & 31;  int t = idx >> 5;
  int h  = t % nh;    t /= nh;
  int l  = t % Lc;    int b = t / Lc;
  const float* row = proj + (size_t)(b * Lc + l) * (nh * HDc) + h * HDc;
  float re = row[2 * pr], im = row[2 * pr + 1];
  float c = fcos[l * (HDc / 2) + pr];
  float s = fsin[l * (HDc / 2) + pr];
  bf16_t* o = out + ((size_t)(b * nh + h) * Lc + l) * HDc;
  o[2 * pr]     = (bf16_t)(re * c - im * s);
  o[2 * pr + 1] = (bf16_t)(re * s + im * c);
}

// vproj: f32 [B*L, KV*64] -> vT: bf16 [B, KV, HD, L]  (transposed per head)
__global__ void v_pack_t_kernel(const float* __restrict__ vproj,
                                bf16_t* __restrict__ out) {
  int idx = blockIdx.x * blockDim.x + threadIdx.x;
  int total = Bc * Lc * KVc * HDc;
  if (idx >= total) return;
  int d = idx & 63;  int t = idx >> 6;
  int h = t % KVc;   t /= KVc;
  int l = t % Lc;    int b = t / Lc;
  out[((size_t)(b * KVc + h) * HDc + d) * Lc + l] =
      (bf16_t)vproj[(size_t)(b * Lc + l) * (KVc * HDc) + h * HDc + d];
}

// ------------------- async global->LDS tile copy (CDNA5) --------------------
// K tile: 32x64 bf16 (4KB), V^T tile: 64x32 bf16 (4KB); 8+8 b128 async loads
// per wave (one 16B chunk per lane per instruction). Tracked by ASYNCcnt.
__device__ __forceinline__ void async_copy_kv(const bf16_t* __restrict__ gK,
                                              const bf16_t* __restrict__ gVT,
                                              const bf16_t* ldsK, const bf16_t* ldsV,
                                              int lane) {
  const uint32_t lk = (uint32_t)(size_t)ldsK;
  const uint32_t lv = (uint32_t)(size_t)ldsV;
#pragma unroll
  for (int c = 0; c < 8; ++c) {
    const int i = c * 32 + lane;                       // 16B chunk index 0..255
    const bf16_t* ga = gK + (i >> 3) * HDc + (i & 7) * 8;
    asm volatile("global_load_async_to_lds_b128 %0, %1, off"
                 :: "v"(lk + i * 16), "v"((uint64_t)(size_t)ga) : "memory");
  }
#pragma unroll
  for (int c = 0; c < 8; ++c) {
    const int i = c * 32 + lane;
    const bf16_t* ga = gVT + (i >> 2) * Lc + (i & 3) * 8;
    asm volatile("global_load_async_to_lds_b128 %0, %1, off"
                 :: "v"(lv + i * 16), "v"((uint64_t)(size_t)ga) : "memory");
  }
}

// --------------------------- flash attention --------------------------------
// One wave per (qtile16, head, batch). Online softmax over 32-key tiles with
// double-buffered async K/V staging in LDS.
__global__ __launch_bounds__(32)
void attn_kernel(const bf16_t* __restrict__ qb, const bf16_t* __restrict__ kb,
                 const bf16_t* __restrict__ vtb, bf16_t* __restrict__ attn) {
  __shared__ __align__(16) bf16_t Klds[2][32 * 64];
  __shared__ __align__(16) bf16_t Vlds[2][64 * 32];
  __shared__ __align__(16) bf16_t plds[16 * 32];
  const int lane = threadIdx.x & 31;
  const int hi   = lane >> 4;
  const int qt = blockIdx.x;            // 0..L/16-1
  const int h  = blockIdx.y;            // 0..H-1
  const int b  = blockIdx.z;            // 0..B-1
  const int kvh = h / Gc;

  const bf16_t* qtile  = qb  + ((size_t)(b * Hc + h)   * Lc + qt * 16) * HDc;
  const bf16_t* kbase  = kb  +  (size_t)(b * KVc + kvh) * Lc * HDc;
  const bf16_t* vtbase = vtb +  (size_t)(b * KVc + kvh) * HDc * Lc;

  const v16bf aq0 = load_a_frag(qtile, HDc);        // d = 0..31
  const v16bf aq1 = load_a_frag(qtile + 32, HDc);   // d = 32..63

  v8f acc0 = {}, acc1 = {}, acc2 = {}, acc3 = {};
  float rowm[8], rows[8];
#pragma unroll
  for (int r = 0; r < 8; ++r) { rowm[r] = -INFINITY; rows[r] = 0.0f; }

  const float scale = 0.125f;                       // 1/sqrt(64)
  const int qmax = qt * 16 + 15;
  const int nkt  = qmax / 32 + 1;

  int buf = 0;
  async_copy_kv(kbase, vtbase, &Klds[0][0], &Vlds[0][0], lane);

  for (int j = 0; j < nkt; ++j) {
    // Prefetch next tile (dummy tile 0 on last iter keeps wait count uniform).
    const int nj = (j + 1 < nkt) ? (j + 1) : 0;
    async_copy_kv(kbase + (size_t)nj * 32 * HDc, vtbase + nj * 32,
                  &Klds[buf ^ 1][0], &Vlds[buf ^ 1][0], lane);
    // 16 chunks of tile j + 16 of tile nj outstanding; async loads complete
    // in order, so <=16 outstanding => tile j fully landed in LDS.
    asm volatile("s_wait_asynccnt 0x10" ::: "memory");

    // ---- S = Q K^T (16x32 scores) via 4 WMMAs; K fragments from LDS ----
    const bf16_t* kl = &Klds[buf][0];
    v16bf bk00 = load_a_frag(kl, HDc);              // keys 0..15,  d 0..31
    v16bf bk01 = load_a_frag(kl + 32, HDc);         // keys 0..15,  d 32..63
    v16bf bk10 = load_a_frag(kl + 16 * HDc, HDc);   // keys 16..31, d 0..31
    v16bf bk11 = load_a_frag(kl + 16 * HDc + 32, HDc);
    v8f s0 = {}, s1 = {};
    s0 = wmma_bf16(aq0, bk00, s0);  s0 = wmma_bf16(aq1, bk01, s0);
    s1 = wmma_bf16(aq0, bk10, s1);  s1 = wmma_bf16(aq1, bk11, s1);

    // ---- mask + online softmax (rows live per 16-lane half) ----
    const int kc0 = j * 32 + (lane & 15);
    const int kc1 = kc0 + 16;
    float p0[8], p1[8], alpha[8];
#pragma unroll
    for (int r = 0; r < 8; ++r) {
      const int qrow = qt * 16 + r + 8 * hi;
      float v0 = (kc0 > qrow) ? -3.0e38f : s0[r] * scale;
      float v1 = (kc1 > qrow) ? -3.0e38f : s1[r] * scale;
      float tm = fmaxf(v0, v1);
#pragma unroll
      for (int off = 1; off < 16; off <<= 1)        // stay inside half-wave
        tm = fmaxf(tm, __shfl_xor(tm, off, 32));
      const float nm = fmaxf(rowm[r], tm);
      alpha[r] = __expf(rowm[r] - nm);
      const float e0 = __expf(v0 - nm);
      const float e1 = __expf(v1 - nm);
      float ts = e0 + e1;
#pragma unroll
      for (int off = 1; off < 16; off <<= 1)
        ts += __shfl_xor(ts, off, 32);
      rows[r] = rows[r] * alpha[r] + ts;
      rowm[r] = nm;
      p0[r] = e0;  p1[r] = e1;
    }
#pragma unroll
    for (int r = 0; r < 8; ++r) {
      acc0[r] *= alpha[r]; acc1[r] *= alpha[r];
      acc2[r] *= alpha[r]; acc3[r] *= alpha[r];
    }

    // ---- C-layout -> A-layout relayout of P through LDS ----
#pragma unroll
    for (int r = 0; r < 8; ++r) {
      const int row = r + 8 * hi;
      plds[row * 32 + (lane & 15)]      = (bf16_t)p0[r];
      plds[row * 32 + 16 + (lane & 15)] = (bf16_t)p1[r];
    }
    __syncthreads();
    const v16bf ap = load_a_frag(plds, 32);
    __syncthreads();

    // ---- O += P V (16x64) via 4 WMMAs; V^T fragments from LDS ----
    const bf16_t* vl = &Vlds[buf][0];
    acc0 = wmma_bf16(ap, load_a_frag(vl,           32), acc0);
    acc1 = wmma_bf16(ap, load_a_frag(vl + 16 * 32, 32), acc1);
    acc2 = wmma_bf16(ap, load_a_frag(vl + 32 * 32, 32), acc2);
    acc3 = wmma_bf16(ap, load_a_frag(vl + 48 * 32, 32), acc3);

    buf ^= 1;
  }

  // ---- normalize and store to attn bf16 [B*L, H*64] ----
#pragma unroll
  for (int r = 0; r < 8; ++r) {
    const float inv = 1.0f / rows[r];
    const int m = r + 8 * hi;
    bf16_t* orow = attn + (size_t)(b * Lc + qt * 16 + m) * (Hc * HDc) + h * HDc;
    orow[(lane & 15)]      = (bf16_t)(acc0[r] * inv);
    orow[16 + (lane & 15)] = (bf16_t)(acc1[r] * inv);
    orow[32 + (lane & 15)] = (bf16_t)(acc2[r] * inv);
    orow[48 + (lane & 15)] = (bf16_t)(acc3[r] * inv);
  }
}

// ------------------------------- launcher -----------------------------------
extern "C" void kernel_launch(void* const* d_in, const int* in_sizes, int n_in,
                              void* d_out, int out_size, void* d_ws, size_t ws_size,
                              hipStream_t stream) {
  (void)in_sizes; (void)n_in; (void)out_size; (void)ws_size;
  const float* x    = (const float*)d_in[0];
  const float* wq   = (const float*)d_in[1];
  const float* wk   = (const float*)d_in[2];
  const float* wv   = (const float*)d_in[3];
  const float* wo   = (const float*)d_in[4];
  const float* fcos = (const float*)d_in[5];
  const float* fsin = (const float*)d_in[6];
  // d_in[7] = mask: causal tril, handled analytically in attn_kernel.
  float* out = (float*)d_out;

  // -------- workspace carve-out (bytes) --------
  char* w = (char*)d_ws;
  size_t off = 0;
  bf16_t* xb    = (bf16_t*)(w + off); off += (size_t)ROWS * DIMc * 2;        // 8 MiB
  bf16_t* wqbT  = (bf16_t*)(w + off); off += (size_t)DIMc * DIMc * 2;        // 2 MiB
  bf16_t* wkbT  = (bf16_t*)(w + off); off += (size_t)DIMc * KVc * HDc * 2;   // .5 MiB
  bf16_t* wvbT  = (bf16_t*)(w + off); off += (size_t)DIMc * KVc * HDc * 2;   // .5 MiB
  bf16_t* wobT  = (bf16_t*)(w + off); off += (size_t)DIMc * DIMc * 2;        // 2 MiB
  float*  qproj = (float*)(w + off);  size_t qproj_off = off;
                                      off += (size_t)ROWS * DIMc * 4;        // 16 MiB
  float*  kproj = (float*)(w + off);  off += (size_t)ROWS * KVc * HDc * 4;   // 4 MiB
  float*  vproj = (float*)(w + off);  off += (size_t)ROWS * KVc * HDc * 4;   // 4 MiB
  bf16_t* qbuf  = (bf16_t*)(w + off); off += (size_t)Bc * Hc  * Lc * HDc * 2;// 8 MiB
  bf16_t* kbuf  = (bf16_t*)(w + off); off += (size_t)Bc * KVc * Lc * HDc * 2;// 2 MiB
  bf16_t* vtbuf = (bf16_t*)(w + off); off += (size_t)Bc * KVc * Lc * HDc * 2;// 2 MiB
  bf16_t* attnb = (bf16_t*)(w + qproj_off);  // alias retired qproj region

  const int T = 256;
  // 1) fp32 -> bf16 conversions (weights transposed for TN GEMM fragments)
  cvt_bf16_kernel  <<<(ROWS * DIMc + T - 1) / T, T, 0, stream>>>(x,  xb,  ROWS * DIMc);
  cvt_bf16_t_kernel<<<(DIMc * DIMc + T - 1) / T, T, 0, stream>>>(wq, wqbT, DIMc, DIMc);
  cvt_bf16_t_kernel<<<(DIMc * 256  + T - 1) / T, T, 0, stream>>>(wk, wkbT, DIMc, 256);
  cvt_bf16_t_kernel<<<(DIMc * 256  + T - 1) / T, T, 0, stream>>>(wv, wvbT, DIMc, 256);
  cvt_bf16_t_kernel<<<(DIMc * DIMc + T - 1) / T, T, 0, stream>>>(wo, wobT, DIMc, DIMc);

  // 2) projections (WMMA GEMMs, TN)
  dim3 gq(DIMc / 64, ROWS / 64);
  gemm_bf16_tn_kernel<<<gq, 128, 0, stream>>>(xb, wqbT, qproj, ROWS, DIMc, DIMc);
  dim3 gkv(256 / 64, ROWS / 64);
  gemm_bf16_tn_kernel<<<gkv, 128, 0, stream>>>(xb, wkbT, kproj, ROWS, 256, DIMc);
  gemm_bf16_tn_kernel<<<gkv, 128, 0, stream>>>(xb, wvbT, vproj, ROWS, 256, DIMc);

  // 3) RoPE + per-head repack (V transposed for contiguous B-fragments)
  rope_pack_kernel<<<(Bc * Lc * Hc  * 32 + T - 1) / T, T, 0, stream>>>(qproj, fcos, fsin, qbuf, Hc);
  rope_pack_kernel<<<(Bc * Lc * KVc * 32 + T - 1) / T, T, 0, stream>>>(kproj, fcos, fsin, kbuf, KVc);
  v_pack_t_kernel <<<(Bc * Lc * KVc * 64 + T - 1) / T, T, 0, stream>>>(vproj, vtbuf);

  // 4) flash attention (async K/V staging, online softmax)
  dim3 ga(Lc / 16, Hc, Bc);
  attn_kernel<<<ga, 32, 0, stream>>>(qbuf, kbuf, vtbuf, attnb);

  // 5) output projection -> d_out (fp32)
  gemm_bf16_tn_kernel<<<gq, 128, 0, stream>>>(attnb, wobT, out, ROWS, DIMc, DIMc);
}
